// DynamicPathCrossAttention_3307124818601
// MI455X (gfx1250) — compile-verified, matrix operand order checked by reading, not run on hardware
//
#include <hip/hip_runtime.h>

typedef unsigned short u16;
typedef unsigned int   u32;
typedef __attribute__((ext_vector_type(16))) __bf16 v16bf;
typedef __attribute__((ext_vector_type(8)))  float  v8f;

union AF { v16bf v; uint4 q[2]; };

__device__ __forceinline__ u16 f2bf(float f) {
    union { float f; u32 u; } x; x.f = f;
    u32 u = x.u;
    u32 r = u + 0x7FFFu + ((u >> 16) & 1u);   // round-to-nearest-even
    return (u16)(r >> 16);
}
__device__ __forceinline__ float bf2f(u16 h) {
    union { u32 u; float f; } x; x.u = ((u32)h) << 16; return x.f;
}

// ---- CDNA5 async global->LDS copy (ASYNCcnt), CDNA5 ISA ----
#define USE_ASYNC_LDS 1

typedef __attribute__((address_space(3))) u16 lds_u16;
__device__ __forceinline__ u32 lds_off(const u16* p) {
    return (u32)(unsigned long long)(lds_u16*)p;   // addrspacecast -> 32-bit LDS byte offset
}
__device__ __forceinline__ void async_b128(u32 ldsByteOff, const void* gaddr) {
#if USE_ASYNC_LDS
    asm volatile("global_load_async_to_lds_b128 %0, %1, off"
                 :: "v"(ldsByteOff), "v"(gaddr) : "memory");
#endif
}
__device__ __forceinline__ void wait_async0() {
#if USE_ASYNC_LDS
    asm volatile("s_wait_asynccnt 0" ::: "memory");
#endif
}

#define BM 128
#define BN 128
#define BK 32
#define LDT 40   // BK + 8 u16 pad: 80B row stride (16B aligned, bank-spread)

// Template params: aBf/bBf = operand precision in memory, bKN = B is [K,N] (else [N,K]),
// OUT: 0 = bf16 store, 1 = f32 store, 2 = f32 atomic-add * gate weight
template<bool aBf, bool bBf, bool bKN, int OUT>
__global__ __launch_bounds__(256) void gemm_wmma(
    const void* __restrict__ Av, const void* __restrict__ Bv,
    const float* __restrict__ bias, void* __restrict__ Cv,
    const float* __restrict__ gate,
    int M, int N, int K, float scale,
    long long sAp, long long sAb, long long sBp, long long sBb,
    long long sCp, long long sCb, long long sBiasP, int gB)
{
    const int z = blockIdx.z;
    const int p = z / gB;
    const int b = z - p * gB;
    float g = 1.0f;
    if (gate) { g = gate[b * 4 + p]; if (g == 0.0f) return; }  // dynamic path skip

    const long long offA = (long long)p * sAp + (long long)b * sAb;
    const long long offB = (long long)p * sBp + (long long)b * sBb;
    const long long offC = (long long)p * sCp + (long long)b * sCb;
    const float* biasP = bias ? bias + (long long)p * sBiasP : nullptr;

    const int n0 = blockIdx.x * BN;
    const int m0 = blockIdx.y * BM;
    const int tid  = threadIdx.x;
    const int wave = tid >> 5, lane = tid & 31;
    const int wm = wave >> 2, wn = wave & 3;   // 2x4 wave grid -> 64x32 output per wave
    const int lm = lane & 15, lh = lane >> 4;

    __shared__ __align__(16) u16 As[2][BM * LDT];
    __shared__ __align__(16) u16 Bs[2][BM * LDT];

    // ---- staging helpers (specialized at compile time) ----
    auto stageA = [&](int buf, int k0) {
        const int r = tid >> 1, c = (tid & 1) << 4;
        u16* dst = &As[buf][r * LDT + c];
        if constexpr (aBf) {
            const u16* s = (const u16*)Av + offA + (long long)(m0 + r) * K + k0 + c;
#if USE_ASYNC_LDS
            const u32 d = lds_off(dst);
            async_b128(d,      s);
            async_b128(d + 16, s + 8);
#else
            *(uint4*)dst       = *(const uint4*)s;
            *(uint4*)(dst + 8) = *(const uint4*)(s + 8);
#endif
        } else {
            const float* s = (const float*)Av + offA + (long long)(m0 + r) * K + k0 + c;
#pragma unroll
            for (int j = 0; j < 16; j += 4) {
                float4 f = *(const float4*)(s + j);
                *(u32*)(dst + j)     = (u32)f2bf(f.x) | ((u32)f2bf(f.y) << 16);
                *(u32*)(dst + j + 2) = (u32)f2bf(f.z) | ((u32)f2bf(f.w) << 16);
            }
        }
    };
    auto stageB = [&](int buf, int k0) {
        if constexpr (!bKN) {                 // B is [N,K] row-major (weights / K-matrix)
            const int r = tid >> 1, c = (tid & 1) << 4;
            u16* dst = &Bs[buf][r * LDT + c];
            if constexpr (bBf) {
                const u16* s = (const u16*)Bv + offB + (long long)(n0 + r) * K + k0 + c;
#if USE_ASYNC_LDS
                const u32 d = lds_off(dst);
                async_b128(d,      s);
                async_b128(d + 16, s + 8);
#else
                *(uint4*)dst       = *(const uint4*)s;
                *(uint4*)(dst + 8) = *(const uint4*)(s + 8);
#endif
            } else {
                const float* s = (const float*)Bv + offB + (long long)(n0 + r) * K + k0 + c;
#pragma unroll
                for (int j = 0; j < 16; j += 4) {
                    float4 f = *(const float4*)(s + j);
                    *(u32*)(dst + j)     = (u32)f2bf(f.x) | ((u32)f2bf(f.y) << 16);
                    *(u32*)(dst + j + 2) = (u32)f2bf(f.z) | ((u32)f2bf(f.w) << 16);
                }
            }
        } else {                              // B is [K,N] row-major bf16 (V): transpose on store
            const int kk = tid >> 3, nn = (tid & 7) << 4;
            const u16* s = (const u16*)Bv + offB + (long long)(k0 + kk) * N + n0 + nn;
            u16 tmp[16];
            *(uint4*)&tmp[0] = *(const uint4*)s;
            *(uint4*)&tmp[8] = *(const uint4*)(s + 8);
#pragma unroll
            for (int j = 0; j < 16; ++j) Bs[buf][(nn + j) * LDT + kk] = tmp[j];
        }
    };
    constexpr bool anyAsync = USE_ASYNC_LDS && (aBf || (bBf && !bKN));

    v8f acc[4][2];
#pragma unroll
    for (int mt = 0; mt < 4; ++mt)
#pragma unroll
        for (int nt = 0; nt < 2; ++nt) {
            v8f zv = {0.f,0.f,0.f,0.f,0.f,0.f,0.f,0.f};
            acc[mt][nt] = zv;
        }

    // ---- software-pipelined K loop: double-buffered LDS, 1 barrier / iter ----
    stageA(0, 0);
    stageB(0, 0);
    if constexpr (anyAsync) wait_async0();
    __syncthreads();

    for (int k0 = 0; k0 < K; k0 += BK) {
        const int cur = (k0 >> 5) & 1;
        const int k1  = k0 + BK;
        if (k1 < K) { stageA(cur ^ 1, k1); stageB(cur ^ 1, k1); }

        // WMMA fragments: lane lm = row/col, lh selects K half (ISA bf16 layout)
        AF fa[4], fb[2];
#pragma unroll
        for (int mt = 0; mt < 4; ++mt) {
            const u16* rp = &As[cur][(wm * 64 + mt * 16 + lm) * LDT];
            fa[mt].q[0] = *(const uint4*)(rp + 8 * lh);        // K = 8*lh .. +7   (V0-3)
            fa[mt].q[1] = *(const uint4*)(rp + 16 + 8 * lh);   // K = 16+8*lh ..+7 (V4-7)
        }
#pragma unroll
        for (int nt = 0; nt < 2; ++nt) {
            const u16* rp = &Bs[cur][(wn * 32 + nt * 16 + lm) * LDT];
            fb[nt].q[0] = *(const uint4*)(rp + 8 * lh);
            fb[nt].q[1] = *(const uint4*)(rp + 16 + 8 * lh);
        }
#pragma unroll
        for (int mt = 0; mt < 4; ++mt)
#pragma unroll
            for (int nt = 0; nt < 2; ++nt)
                acc[mt][nt] = __builtin_amdgcn_wmma_f32_16x16x32_bf16(
                    false, fa[mt].v, false, fb[nt].v, (short)0, acc[mt][nt], false, false);

        if (k1 < K) {
            if constexpr (anyAsync) wait_async0();
            __syncthreads();
        }
    }

    // ---- epilogue: C layout lane = n, VGPR v + 8*lh = m within tile ----
#pragma unroll
    for (int mt = 0; mt < 4; ++mt) {
#pragma unroll
        for (int nt = 0; nt < 2; ++nt) {
            const int n = n0 + wn * 32 + nt * 16 + lm;
            const float bv2 = biasP ? biasP[n] : 0.0f;
#pragma unroll
            for (int v = 0; v < 8; ++v) {
                const int m = m0 + wm * 64 + mt * 16 + 8 * lh + v;
                const float cv = acc[mt][nt][v] * scale + bv2;
                const long long ci = offC + (long long)m * N + n;
                if constexpr (OUT == 2) {
                    (void)__hip_atomic_fetch_add((float*)Cv + ci, cv * g,
                                                 __ATOMIC_RELAXED, __HIP_MEMORY_SCOPE_AGENT);
                } else if constexpr (OUT == 1) {
                    ((float*)Cv)[ci] = cv;
                } else {
                    ((u16*)Cv)[ci] = f2bf(cv);
                }
            }
        }
    }
}

__global__ void zero_f32(float* __restrict__ p, int n) {
    int i = blockIdx.x * 256 + threadIdx.x;
    if (i < n) p[i] = 0.0f;
}

// pooled[b][e] = mean_q Qp[b][q][e]
__global__ void pool_kernel(const u16* __restrict__ qp, float* __restrict__ pooled) {
    const int t = blockIdx.x * 256 + threadIdx.x;     // t in [0, 8*1024)
    const int b = t >> 10, e = t & 1023;
    const u16* src = qp + (long long)b * 1024 * 1024 + e;
    float s = 0.0f;
    for (int q = 0; q < 1024; ++q) s += bf2f(src[(long long)q * 1024]);
    pooled[t] = s * (1.0f / 1024.0f);
}

// h = relu(pooled @ Wm1^T + bm1); logits = h @ Wm2^T + bm2; softmax; top-2 renorm -> gate[B][P]
__global__ void gate_kernel(const float* __restrict__ pooled,
                            const float* __restrict__ Wm1, const float* __restrict__ bm1,
                            const float* __restrict__ Wm2, const float* __restrict__ bm2,
                            float* __restrict__ gate) {
    __shared__ float hS[8 * 1024];
    __shared__ float lg[32];
    const int tid = threadIdx.x;
    for (int idx = tid; idx < 8 * 1024; idx += 256) {
        const int b = idx >> 10, e = idx & 1023;
        const float* x = pooled + b * 1024;
        const float* w = Wm1 + (long long)e * 1024;
        float s = bm1[e];
        for (int d = 0; d < 1024; ++d) s += x[d] * w[d];
        hS[idx] = fmaxf(s, 0.0f);
    }
    __syncthreads();
    if (tid < 32) {
        const int b = tid >> 2, pp = tid & 3;
        const float* h = hS + b * 1024;
        const float* w = Wm2 + pp * 1024;
        float s = bm2[pp];
        for (int d = 0; d < 1024; ++d) s += h[d] * w[d];
        lg[tid] = s;
    }
    __syncthreads();
    if (tid < 8) {
        float pr[4];
        float m = lg[tid * 4];
        for (int j = 1; j < 4; ++j) m = fmaxf(m, lg[tid * 4 + j]);
        float s = 0.0f;
        for (int j = 0; j < 4; ++j) { pr[j] = __expf(lg[tid * 4 + j] - m); s += pr[j]; }
        for (int j = 0; j < 4; ++j) pr[j] /= s;
        int i1 = 0;
        for (int j = 1; j < 4; ++j) if (pr[j] > pr[i1]) i1 = j;
        int i2 = -1;
        for (int j = 0; j < 4; ++j) if (j != i1 && (i2 < 0 || pr[j] > pr[i2])) i2 = j;
        const float den = pr[i1] + pr[i2] + 1e-8f;
        for (int j = 0; j < 4; ++j)
            gate[tid * 4 + j] = (j == i1 || j == i2) ? pr[j] / den : 0.0f;
    }
}

// in-place row softmax on bf16 logits [P][B][LQ][LK]; skips gated-off (p,b)
__global__ void softmax_rows(u16* __restrict__ S, const float* __restrict__ gate) {
    const int row = blockIdx.x;                 // [0, 32*1024)
    const int p = row >> 13;                    // 8192 rows per path
    const int b = (row >> 10) & 7;
    if (gate[b * 4 + p] == 0.0f) return;
    u16* r = S + (long long)row * 1024;
    const int tid = threadIdx.x;
    __shared__ float red[256];
    float x[4];
#pragma unroll
    for (int j = 0; j < 4; ++j) x[j] = bf2f(r[tid + j * 256]);
    float m = fmaxf(fmaxf(x[0], x[1]), fmaxf(x[2], x[3]));
    red[tid] = m; __syncthreads();
    for (int s2 = 128; s2 > 0; s2 >>= 1) {
        if (tid < s2) red[tid] = fmaxf(red[tid], red[tid + s2]);
        __syncthreads();
    }
    m = red[0]; __syncthreads();
    float e[4]; float sum = 0.0f;
#pragma unroll
    for (int j = 0; j < 4; ++j) { e[j] = __expf(x[j] - m); sum += e[j]; }
    red[tid] = sum; __syncthreads();
    for (int s2 = 128; s2 > 0; s2 >>= 1) {
        if (tid < s2) red[tid] += red[tid + s2];
        __syncthreads();
    }
    const float inv = 1.0f / red[0];
#pragma unroll
    for (int j = 0; j < 4; ++j) r[tid + j * 256] = f2bf(e[j] * inv);
}

extern "C" void kernel_launch(void* const* d_in, const int* in_sizes, int n_in,
                              void* d_out, int out_size, void* d_ws, size_t ws_size,
                              hipStream_t stream) {
    const float* Q   = (const float*)d_in[0];
    const float* src = (const float*)d_in[1];
    const float* Wq  = (const float*)d_in[2];
    const float* bq  = (const float*)d_in[3];
    const float* Wk  = (const float*)d_in[4];
    const float* bk  = (const float*)d_in[5];
    const float* Wv  = (const float*)d_in[6];
    const float* bv  = (const float*)d_in[7];
    const float* Wm1 = (const float*)d_in[8];
    const float* bm1 = (const float*)d_in[9];
    const float* Wm2 = (const float*)d_in[10];
    const float* bm2 = (const float*)d_in[11];
    const float* Wo  = (const float*)d_in[12];
    const float* bo  = (const float*)d_in[13];
    float* out = (float*)d_out;
    (void)in_sizes; (void)n_in; (void)out_size; (void)ws_size;

    char* ws = (char*)d_ws;
    size_t off = 0;
    auto take = [&](size_t bytes) -> char* {
        off = (off + 255) & ~(size_t)255;
        char* r = ws + off; off += bytes; return r;
    };
    const long long E1M = 1024LL * 1024;       // elements per [1024,1024] matrix
    const long long E8M = 8LL * 1024 * 1024;

    u16*   qp     = (u16*)take(E8M * 2);        // Qp   bf16 [B*LQ, D]       16 MB
    u16*   kb     = (u16*)take(4 * E8M * 2);    // K    bf16 [P][B][LK][D]   64 MB
    u16*   vb     = (u16*)take(4 * E8M * 2);    // V    bf16                 64 MB
    u16*   sb     = (u16*)take(4 * E8M * 2);    // attn bf16 [P][B][LQ][LK]  64 MB
    float* pooled = (float*)take(8 * 1024 * 4);
    float* gatep  = (float*)take(32 * 4);
    float* oacc   = (float*)take(E8M * 4);      // sum_p w*attn_out, f32     32 MB

    const dim3 blk(256);

    // 0) zero the path-combine accumulator (atomic target)
    zero_f32<<<dim3(32768), blk, 0, stream>>>(oacc, 8 * 1024 * 1024);

    // 1) Qp = Q @ Wq^T + bq  -> bf16            M=8192 N=1024 K=1024
    gemm_wmma<false,false,false,0><<<dim3(8, 64, 1), blk, 0, stream>>>(Q, Wq, bq, qp, nullptr,
        8192, 1024, 1024, 1.0f, 0, 0, 0, 0, 0, 0, 0, 1);

    // 2) K[p] = src[p] @ Wk[p]^T + bk[p] -> bf16   (batched over p)
    gemm_wmma<false,false,false,0><<<dim3(8, 64, 4), blk, 0, stream>>>(src, Wk, bk, kb, nullptr,
        8192, 1024, 1024, 1.0f, E8M, 0, E1M, 0, E8M, 0, 1024, 1);

    // 3) V[p] likewise
    gemm_wmma<false,false,false,0><<<dim3(8, 64, 4), blk, 0, stream>>>(src, Wv, bv, vb, nullptr,
        8192, 1024, 1024, 1.0f, E8M, 0, E1M, 0, E8M, 0, 1024, 1);

    // 4) pooled mean + gating MLP -> sparse top-2 path weights
    pool_kernel<<<dim3(32), blk, 0, stream>>>(qp, pooled);
    gate_kernel<<<dim3(1), blk, 0, stream>>>(pooled, Wm1, bm1, Wm2, bm2, gatep);

    // 5) logits[p,b] = (Qp[b] @ K[p,b]^T) / 32 -> bf16 (skip gated-off paths; async LDS staging)
    gemm_wmma<true,true,false,0><<<dim3(8, 8, 32), blk, 0, stream>>>(qp, kb, nullptr, sb, gatep,
        1024, 1024, 1024, 0.03125f,
        0, E1M, E8M, E1M, E8M, E1M, 0, 8);

    // 6) softmax rows in place
    softmax_rows<<<dim3(32768), blk, 0, stream>>>(sb, gatep);

    // 7) oacc[b] += gate[b,p] * (attn[p,b] @ V[p,b])   (B is [K,N], atomic f32)
    gemm_wmma<true,true,true,2><<<dim3(8, 8, 32), blk, 0, stream>>>(sb, vb, nullptr, oacc, gatep,
        1024, 1024, 1024, 1.0f,
        E8M, E1M, E8M, E1M, 0, E1M, 0, 8);

    // 8) out = oacc @ Wo^T + bo -> f32 d_out
    gemm_wmma<false,false,false,1><<<dim3(8, 64, 1), blk, 0, stream>>>(oacc, Wo, bo, out, nullptr,
        8192, 1024, 1024, 1.0f, 0, 0, 0, 0, 0, 0, 0, 1);
}
